// StyleGAN2Block_50929722196862
// MI455X (gfx1250) — matrix operand order repaired and checked by reading, not compiled, not run on hardware
//
#include <hip/hip_runtime.h>
#include <hip/hip_bf16.h>
#include <cmath>

// ---------------------------------------------------------------------------
// StyleGAN2 block for MI455X (gfx1250, wave32, WMMA).
//   B=4, C=512, H=W=64 (after 2x upsample), K=3.
// Modulation folded into input scaling, demodulation into output scaling ->
// both 3x3 convs are batch-shared-weight implicit GEMMs on the matrix cores
// (v_wmma_f32_16x16x32_bf16), M=512, K=4608, N=16384.
// ---------------------------------------------------------------------------

#define BSZ   4
#define CH    512
#define HW    64
#define HWIN  32

typedef __attribute__((ext_vector_type(16))) __bf16        v16bf;
typedef __attribute__((ext_vector_type(8)))  float         v8f;
typedef __attribute__((ext_vector_type(4)))  unsigned int  u32x4;

union FragU { v16bf bf; u32x4 q[2]; };

__device__ __forceinline__ unsigned short f32_to_bf16(float f) {
    unsigned u = __float_as_uint(f);
    u += 0x7FFFu + ((u >> 16) & 1u);          // round-to-nearest-even
    return (unsigned short)(u >> 16);
}
__device__ __forceinline__ float bf16_to_f32(unsigned short h) {
    return __uint_as_float(((unsigned)h) << 16);
}

// bilinear 2x (half-pixel centers, edge clamp): out o from in of length 32
__device__ __forceinline__ void up2_taps(int o, int& i0, int& i1, float& w0, float& w1) {
    int m = o >> 1;
    if (o & 1) { i0 = m; i1 = (m + 1 < HWIN) ? m + 1 : HWIN - 1; w0 = 0.75f; w1 = 0.25f; }
    else       { i1 = m; i0 = (m - 1 >= 0) ? m - 1 : 0;          w0 = 0.25f; w1 = 0.75f; }
}

// ---------------------------------------------------------------------------
// s[b][c] = (w[b] . ms_w[c]) * MS_SCALE + ms_b[c]
__global__ __launch_bounds__(256)
void style_kernel(const float* __restrict__ w, const float* __restrict__ msw,
                  const float* __restrict__ msb, float ms_scale, float* __restrict__ s) {
    int i = blockIdx.x * 256 + threadIdx.x;          // b*512 + c
    if (i >= BSZ * CH) return;
    int b = i >> 9, c = i & 511;
    const float* wr = w + (size_t)b * CH;
    const float* mr = msw + (size_t)c * CH;
    float acc = 0.f;
    for (int k = 0; k < CH; ++k) acc += wr[k] * mr[k];
    s[i] = acc * ms_scale + msb[c];
}

// wsq[cout][cin] = sum over 3x3 of W^2
__global__ __launch_bounds__(256)
void wsq_kernel(const float* __restrict__ W, float* __restrict__ wsq) {
    int i = blockIdx.x * 256 + threadIdx.x;          // cout*512 + cin
    if (i >= CH * CH) return;
    const float* p = W + (size_t)i * 9;
    float a = 0.f;
#pragma unroll
    for (int k = 0; k < 9; ++k) { float v = p[k]; a += v * v; }
    wsq[i] = a;
}

// d[b][cout] = rsqrt(scale^2 * sum_cin wsq[cout][cin]*s[b][cin]^2 + 1e-8)
__global__ __launch_bounds__(256)
void demod_kernel(const float* __restrict__ wsq, const float* __restrict__ s,
                  float conv_scale, float* __restrict__ d) {
    int i = blockIdx.x * 256 + threadIdx.x;          // b*512 + cout
    if (i >= BSZ * CH) return;
    int b = i >> 9, co = i & 511;
    const float* wr = wsq + (size_t)co * CH;
    const float* sr = s + (size_t)b * CH;
    float a = 0.f;
    for (int k = 0; k < CH; ++k) { float sv = sr[k]; a += wr[k] * sv * sv; }
    d[i] = rsqrtf(a * conv_scale * conv_scale + 1e-8f);
}

// bf16 weights, layout [tap(9)][cout][cin], scaled by CONV_SCALE
__global__ __launch_bounds__(256)
void wconv_kernel(const float* __restrict__ W, float conv_scale,
                  unsigned short* __restrict__ wb) {
    int i = blockIdx.x * 256 + threadIdx.x;          // ((pos*512)+cout)*512+cin
    if (i >= 9 * CH * CH) return;
    int cin = i & 511; int t = i >> 9; int cout = t & 511; int pos = t >> 9;
    float v = W[((size_t)cout * CH + cin) * 9 + pos] * conv_scale;
    wb[i] = f32_to_bf16(v);
}

// upsample maps 32->64 bilinear, scale by s0[b][c], write NHWC bf16
__global__ __launch_bounds__(512)
void upsample_kernel(const float* __restrict__ maps, const float* __restrict__ s0,
                     unsigned short* __restrict__ xs) {
    int c = threadIdx.x;
    int x = blockIdx.x, y = blockIdx.y, b = blockIdx.z;
    int y0, y1, x0, x1; float wy0, wy1, wx0, wx1;
    up2_taps(y, y0, y1, wy0, wy1);
    up2_taps(x, x0, x1, wx0, wx1);
    const float* p = maps + ((size_t)b * CH + c) * (HWIN * HWIN);
    float v = wy0 * (wx0 * p[y0 * HWIN + x0] + wx1 * p[y0 * HWIN + x1]) +
              wy1 * (wx0 * p[y1 * HWIN + x0] + wx1 * p[y1 * HWIN + x1]);
    xs[(((size_t)b * HW + y) * HW + x) * CH + c] = f32_to_bf16(v * s0[b * CH + c]);
}

// ---------------------------------------------------------------------------
// Implicit-GEMM 3x3 conv via WMMA bf16, software-pipelined.
// block = 256 threads (8 waves). One block: (b, y-row) x 128 couts x 64 pixels.
// Each wave: 16 couts x 64 pixels = 4 accumulator tiles; each A (weight)
// fragment feeds 4 WMMAs. Input row-tile (3 x 66 x 32cin, bf16) staged in LDS;
// next cin-chunk is prefetched into registers while the current one computes.
#define TCOLS 66                       // x = -1 .. 64
#define NSEG  (3 * TCOLS * 4)          // 16B segments per chunk = 792
__global__ __launch_bounds__(256)
void conv3x3_wmma_kernel(const unsigned short* __restrict__ xs,     // [B][64][64][512] bf16 (pre-scaled by s)
                         const unsigned short* __restrict__ wb,     // [9][512][512] bf16 (pre-scaled)
                         const float* __restrict__ demod,           // [B][512]
                         const float* __restrict__ bias,            // [512]
                         const float* __restrict__ ns,              // [512]
                         const float* __restrict__ noise,           // [B][64][64]
                         const float* __restrict__ snext,           // [B][512] scale for next layer input
                         unsigned short* __restrict__ xnext,        // [B][64][64][512] bf16 = h*snext
                         float* __restrict__ outh)                  // [B][512][64][64] f32 or null
{
    __shared__ __align__(64) unsigned short lds[3 * TCOLS * 32];    // 12672 B

    const int tid  = threadIdx.x;
    const int wave = tid >> 5;
    const int lane = tid & 31;
    const int lh   = lane >> 4;        // lane half (0: lanes 0-15, 1: lanes 16-31)
    const int n    = lane & 15;        // pixel col within N-tile / M row for A

    const int y  = blockIdx.y;
    const int b  = blockIdx.z;
    const int coutBase = blockIdx.x * 128 + wave * 16;

    // --- per-thread staging descriptors (cin-chunk invariant) ---
    const unsigned short* gsrc[4];
    unsigned short*       ldst[4];
    bool                  gval[4];
    bool                  gact[4];
#pragma unroll
    for (int k = 0; k < 4; ++k) {
        int s   = tid + k * 256;
        bool act = (s < NSEG);
        int ss   = act ? s : 0;
        int pair = ss >> 2;            // (row,col) 0..197
        int seg  = ss & 3;             // 8-ushort segment within 32 cin
        int r    = pair / TCOLS;
        int cc   = pair - r * TCOLS;
        int yy   = y + r - 1;
        int xx   = cc - 1;
        gact[k] = act;
        gval[k] = act && (unsigned)yy < (unsigned)HW && (unsigned)xx < (unsigned)HW;
        gsrc[k] = xs + (((size_t)b * HW + (yy < 0 ? 0 : yy)) * HW + (xx < 0 ? 0 : xx)) * CH + seg * 8;
        ldst[k] = lds + pair * 32 + seg * 8;
    }

    // per-wave compute base pointers
    const unsigned short* lp = lds + (n * 32 + lh * 8);                     // B frags
    const unsigned short* wp = wb + ((size_t)(coutBase + n)) * CH + lh * 8; // A frags

    v8f acc0 = {0.f,0.f,0.f,0.f,0.f,0.f,0.f,0.f};
    v8f acc1 = acc0, acc2 = acc0, acc3 = acc0;

    // prefetch chunk 0
    u32x4 pf[4];
    const u32x4 zero4 = {0u, 0u, 0u, 0u};
#pragma unroll
    for (int k = 0; k < 4; ++k)
        pf[k] = gval[k] ? *(const u32x4*)(gsrc[k]) : zero4;

    for (int cb = 0; cb < CH; cb += 32) {
        __syncthreads();                       // LDS free from previous compute
#pragma unroll
        for (int k = 0; k < 4; ++k)
            if (gact[k]) *(u32x4*)(ldst[k]) = pf[k];
        __syncthreads();                       // tile visible to all waves

        if (cb + 32 < CH) {                    // prefetch next chunk (overlaps compute)
#pragma unroll
            for (int k = 0; k < 4; ++k)
                pf[k] = gval[k] ? *(const u32x4*)(gsrc[k] + cb + 32) : zero4;
        }

        const unsigned short* wpc = wp + cb;
#pragma unroll
        for (int pos = 0; pos < 9; ++pos) {
            const int ky = pos / 3, kx = pos - ky * 3;
            FragU fa;                          // one A fragment, reused by 4 tiles
            fa.q[0] = *(const u32x4*)(wpc + pos * (CH * CH));
            fa.q[1] = *(const u32x4*)(wpc + pos * (CH * CH) + 16);

            FragU fb0, fb1, fb2, fb3;
            const unsigned short* bp = lp + (ky * TCOLS + kx) * 32;
            fb0.q[0] = *(const u32x4*)(bp);
            fb0.q[1] = *(const u32x4*)(bp + 16);
            fb1.q[0] = *(const u32x4*)(bp + 16 * 32);
            fb1.q[1] = *(const u32x4*)(bp + 16 * 32 + 16);
            fb2.q[0] = *(const u32x4*)(bp + 32 * 32);
            fb2.q[1] = *(const u32x4*)(bp + 32 * 32 + 16);
            fb3.q[0] = *(const u32x4*)(bp + 48 * 32);
            fb3.q[1] = *(const u32x4*)(bp + 48 * 32 + 16);

            acc0 = __builtin_amdgcn_wmma_f32_16x16x32_bf16(false, fa.bf, false, fb0.bf, (short)0, acc0, false, false);
            acc1 = __builtin_amdgcn_wmma_f32_16x16x32_bf16(false, fa.bf, false, fb1.bf, (short)0, acc1, false, false);
            acc2 = __builtin_amdgcn_wmma_f32_16x16x32_bf16(false, fa.bf, false, fb2.bf, (short)0, acc2, false, false);
            acc3 = __builtin_amdgcn_wmma_f32_16x16x32_bf16(false, fa.bf, false, fb3.bf, (short)0, acc3, false, false);
        }
    }

    // --- epilogue: D layout — VGPR r, lane -> (M = r + 8*lh, N = n) ---
    float dmv[8], bsv[8], nsv[8];
#pragma unroll
    for (int r = 0; r < 8; ++r) {
        const int cout = coutBase + r + lh * 8;
        dmv[r] = demod[b * CH + cout];
        bsv[r] = bias[cout] ;
        nsv[r] = ns[cout];
    }
    const float snv_base = 0.f; (void)snv_base;
#pragma unroll
    for (int t = 0; t < 4; ++t) {
        const v8f accv = (t == 0) ? acc0 : (t == 1) ? acc1 : (t == 2) ? acc2 : acc3;
        const int x = t * 16 + n;
        const float nz = noise[((size_t)b * HW + y) * HW + x];
#pragma unroll
        for (int r = 0; r < 8; ++r) {
            const int cout = coutBase + r + lh * 8;
            float v = accv[r] * dmv[r] + bsv[r] + nsv[r] * nz;
            float h = (v >= 0.f) ? v : 0.2f * v;
            if (outh)
                outh[(((size_t)b * CH + cout) * HW + y) * HW + x] = h;
            xnext[(((size_t)b * HW + y) * HW + x) * CH + cout] =
                f32_to_bf16(h * snext[b * CH + cout]);
        }
    }
}

// ---------------------------------------------------------------------------
// to_rgb: 1x1 modulated conv (no demod) + upsampled skip rgb.
// One wave per output pixel; lane-strided K reduction + shuffle reduce.
__global__ __launch_bounds__(256)
void torgb_kernel(const unsigned short* __restrict__ xsr,   // [B][64][64][512] bf16 = h*sr
                  const float* __restrict__ rgbw,           // [3][512]
                  const float* __restrict__ rgbb,           // [3]
                  const float* __restrict__ rgbin,          // [B][3][32][32]
                  float rgb_scale,
                  float* __restrict__ out)                  // [B][3][64][64]
{
    const int wave = threadIdx.x >> 5;
    const int lane = threadIdx.x & 31;
    const int p = blockIdx.x * 8 + wave;                    // pixel id 0..B*4096-1
    const int b = p >> 12;
    const int yx = p & 4095;
    const int y = yx >> 6, x = yx & 63;

    const unsigned short* xp = xsr + (size_t)p * CH;
    float a0 = 0.f, a1 = 0.f, a2 = 0.f;
    for (int c = lane; c < CH; c += 32) {
        float xv = bf16_to_f32(xp[c]);
        a0 += xv * rgbw[c];
        a1 += xv * rgbw[CH + c];
        a2 += xv * rgbw[2 * CH + c];
    }
#pragma unroll
    for (int off = 16; off > 0; off >>= 1) {
        a0 += __shfl_down(a0, off, 32);
        a1 += __shfl_down(a1, off, 32);
        a2 += __shfl_down(a2, off, 32);
    }
    if (lane == 0) {
        int y0, y1, x0, x1; float wy0, wy1, wx0, wx1;
        up2_taps(y, y0, y1, wy0, wy1);
        up2_taps(x, x0, x1, wx0, wx1);
        float a[3] = {a0, a1, a2};
#pragma unroll
        for (int o = 0; o < 3; ++o) {
            const float* rp = rgbin + ((size_t)b * 3 + o) * (HWIN * HWIN);
            float up = wy0 * (wx0 * rp[y0 * HWIN + x0] + wx1 * rp[y0 * HWIN + x1]) +
                       wy1 * (wx0 * rp[y1 * HWIN + x0] + wx1 * rp[y1 * HWIN + x1]);
            out[((size_t)b * 3 + o) * (HW * HW) + yx] = up + rgb_scale * a[o] + rgbb[o];
        }
    }
}

// ---------------------------------------------------------------------------
extern "C" void kernel_launch(void* const* d_in, const int* in_sizes, int n_in,
                              void* d_out, int out_size, void* d_ws, size_t ws_size,
                              hipStream_t stream) {
    (void)in_sizes; (void)n_in; (void)out_size; (void)ws_size;

    const float* maps    = (const float*)d_in[0];
    const float* w       = (const float*)d_in[1];
    const float* rgb     = (const float*)d_in[2];
    const float* noise0  = (const float*)d_in[3];
    const float* noise1  = (const float*)d_in[4];
    const float* conv0_w = (const float*)d_in[5];
    const float* conv0_b = (const float*)d_in[6];
    const float* ms0_w   = (const float*)d_in[7];
    const float* ms0_b   = (const float*)d_in[8];
    const float* ns0     = (const float*)d_in[9];
    const float* conv1_w = (const float*)d_in[10];
    const float* conv1_b = (const float*)d_in[11];
    const float* ms1_w   = (const float*)d_in[12];
    const float* ms1_b   = (const float*)d_in[13];
    const float* ns1     = (const float*)d_in[14];
    const float* rgb_w   = (const float*)d_in[15];
    const float* rgb_b   = (const float*)d_in[16];
    const float* msr_w   = (const float*)d_in[17];
    const float* msr_b   = (const float*)d_in[18];

    float* out_rgb = (float*)d_out;                      // B*3*64*64
    float* out_h   = (float*)d_out + BSZ * 3 * HW * HW;  // B*512*64*64

    const double A = 0.2;
    const float G          = (float)sqrt(2.0 / (1.0 + A * A));
    const float CONV_SCALE = G / (float)sqrt((double)(CH * 9));
    const float MS_SCALE   = (float)sqrt(2.0 / (double)(CH + CH));
    const float RGB_SCALE  = G * (float)sqrt(2.0 / (double)(CH + 3));

    // ---- workspace carve-up (256B aligned) ----
    char* ws = (char*)d_ws;
    size_t off = 0;
    auto alloc = [&](size_t bytes) {
        char* p = ws + off;
        off += (bytes + 255) & ~(size_t)255;
        return p;
    };
    float*          s0   = (float*)alloc(BSZ * CH * 4);
    float*          s1   = (float*)alloc(BSZ * CH * 4);
    float*          sr   = (float*)alloc(BSZ * CH * 4);
    float*          dm0  = (float*)alloc(BSZ * CH * 4);
    float*          dm1  = (float*)alloc(BSZ * CH * 4);
    float*          wsq0 = (float*)alloc((size_t)CH * CH * 4);
    float*          wsq1 = (float*)alloc((size_t)CH * CH * 4);
    unsigned short* wb0  = (unsigned short*)alloc((size_t)9 * CH * CH * 2);
    unsigned short* wb1  = (unsigned short*)alloc((size_t)9 * CH * CH * 2);
    unsigned short* xs0  = (unsigned short*)alloc((size_t)BSZ * HW * HW * CH * 2);
    unsigned short* xs1  = (unsigned short*)alloc((size_t)BSZ * HW * HW * CH * 2);
    unsigned short* xsr  = (unsigned short*)alloc((size_t)BSZ * HW * HW * CH * 2);

    // ---- small prep kernels ----
    style_kernel<<<(BSZ * CH + 255) / 256, 256, 0, stream>>>(w, ms0_w, ms0_b, MS_SCALE, s0);
    style_kernel<<<(BSZ * CH + 255) / 256, 256, 0, stream>>>(w, ms1_w, ms1_b, MS_SCALE, s1);
    style_kernel<<<(BSZ * CH + 255) / 256, 256, 0, stream>>>(w, msr_w, msr_b, MS_SCALE, sr);

    wsq_kernel<<<(CH * CH + 255) / 256, 256, 0, stream>>>(conv0_w, wsq0);
    wsq_kernel<<<(CH * CH + 255) / 256, 256, 0, stream>>>(conv1_w, wsq1);

    demod_kernel<<<(BSZ * CH + 255) / 256, 256, 0, stream>>>(wsq0, s0, CONV_SCALE, dm0);
    demod_kernel<<<(BSZ * CH + 255) / 256, 256, 0, stream>>>(wsq1, s1, CONV_SCALE, dm1);

    wconv_kernel<<<(9 * CH * CH + 255) / 256, 256, 0, stream>>>(conv0_w, CONV_SCALE, wb0);
    wconv_kernel<<<(9 * CH * CH + 255) / 256, 256, 0, stream>>>(conv1_w, CONV_SCALE, wb1);

    // ---- upsample + modulate maps -> xs0 ----
    upsample_kernel<<<dim3(HW, HW, BSZ), 512, 0, stream>>>(maps, s0, xs0);

    // ---- layer 0: conv -> noise -> lrelu, emit xs1 = h*s1 ----
    conv3x3_wmma_kernel<<<dim3(4, HW, BSZ), 256, 0, stream>>>(
        xs0, wb0, dm0, conv0_b, ns0, noise0, s1, xs1, nullptr);

    // ---- layer 1: conv -> noise -> lrelu, emit h (f32 NCHW) and xsr = h*sr ----
    conv3x3_wmma_kernel<<<dim3(4, HW, BSZ), 256, 0, stream>>>(
        xs1, wb1, dm1, conv1_b, ns1, noise1, sr, xsr, out_h);

    // ---- to_rgb + upsampled skip ----
    torgb_kernel<<<(BSZ * HW * HW) / 8, 256, 0, stream>>>(
        xsr, rgb_w, rgb_b, rgb, RGB_SCALE, out_rgb);
}
